// AutoRegressiveGraphConvLayer_53386443489592
// MI455X (gfx1250) — matrix operand order repaired and checked by reading, not compiled, not run on hardware
//
#include <hip/hip_runtime.h>

// AutoRegressiveGraphConvLayer fused kernel for MI455X (gfx1250, wave32).
// v3: software-pipelined (double-buffered) LDS fragment loads so each
// v_wmma_f32_16x16x32_f16 waits only on fragments loaded one step ahead
// (s_wait_dscnt > 0) instead of stalling on its own just-issued loads.
// Tile = one node x a batch pair (M=128 rows); A-frag feeds 8/4 WMMAs.

typedef _Float16 half_t;
typedef __attribute__((ext_vector_type(16))) _Float16 v16h;
typedef __attribute__((ext_vector_type(8)))  float    v8f;

#define DEVINL __device__ __forceinline__

constexpr int       Bc    = 8;
constexpr int       Nn    = 1024;
constexpr int       FN    = 64;
constexpr long long E_TOT = 63456;   // sum_{i<1024} min(i,64)

// LDS strides (halves / floats); row strides stay multiples of 8 halves (16B)
constexpr int SXN = 200;  // A-tile row stride (192 cols + 8 pad)
constexpr int SW1 = 200;  // Wn1^T row stride (K=192)
constexpr int SW  = 136;  // K=128 weight row stride
constexpr int SH  = 136;  // hidden tile stride
constexpr int SPE = 68;   // pair_e f32 tile stride

// f16 weight arena (halves), all matrices stored transposed [N][K+pad]
constexpr int OFF_WN1 = 0;                    // 128 x SW1
constexpr int OFF_WN2 = OFF_WN1 + 128 * SW1;  //  64 x SW
constexpr int OFF_WE1 = OFF_WN2 + 64 * SW;    // 128 x SW
constexpr int OFF_WE2 = OFF_WE1 + 128 * SW;   //  64 x SW
constexpr int OFF_WEO = OFF_WE2 + 64 * SW;    //  64 x SW  (W_edges)
constexpr int OFF_WNO = OFF_WEO + 64 * SW;    //  64 x SW  (W_nodes)
constexpr int W_HALVES = OFF_WNO + 64 * SW;   // 77824 halves = 152KB

// Dynamic-LDS byte offsets (tile has 128 rows: 2 batches x 64)
constexpr int LB_W   = 0;
constexpr int LB_XN  = LB_W + W_HALVES * 2;       // 128 x SXN halves
constexpr int LB_H   = LB_XN + 128 * SXN * 2;     // 128 x SH halves
constexpr int LB_PE  = LB_H + 128 * SH * 2;       // 128 x SPE floats
constexpr int LB_RED = LB_PE + 128 * SPE * 4;     // [16][64] partials
constexpr int LB_AGG = LB_RED + 16 * 64 * 4;      // [2][64]
constexpr int LB_B1N = LB_AGG + 2 * 64 * 4;
constexpr int LB_B1E = LB_B1N + 128 * 4;
constexpr int LB_B2N = LB_B1E + 128 * 4;
constexpr int LB_B2E = LB_B2N + 64 * 4;
constexpr int LB_BO  = LB_B2E + 64 * 4;
constexpr int LDS_BYTES = LB_BO + 64 * 4;         // ~276KB < 320KB/WGP

// ---------------- prologue: fp32 [K][N] -> f16 transposed [N][K+pad] --------
__global__ void wcvt_kernel(const float* __restrict__ src, half_t* __restrict__ dst,
                            int K, int N, int stride) {
  int i = blockIdx.x * blockDim.x + threadIdx.x;
  int total = N * stride;
  if (i >= total) return;
  int n = i / stride, k = i % stride;
  float v = (k < K) ? src[(long long)k * N + n] : 0.0f;
  dst[(long long)n * stride + k] = (half_t)v;
}

// ---------------- WMMA fragment helpers (ISA 7.12.2 layouts) ----------------
union Frag { v16h h; uint4 q[2]; };

// A (16x32 f16): lane -> row M=lane&15, g=lane>>4. Halves held are the two
// contiguous 16B runs [8g, 8g+8) and [16+8g, 24+8g)  -> 2x ds_load_b128.
DEVINL void load_a_frag(Frag& f, const half_t* p, int row0, int col0, int stride, int lane) {
  const int m = lane & 15, g = lane >> 4;
  const half_t* base = p + (row0 + m) * stride + col0 + 8 * g;
  f.q[0] = *reinterpret_cast<const uint4*>(base);
  f.q[1] = *reinterpret_cast<const uint4*>(base + 16);
}

// B (32x16 f16) from transposed weights [n][k]: lane -> col n=lane&15,
// g=lane>>4; halves are the two runs [8g, 8g+8) and [16+8g, 24+8g) -> 2x b128.
DEVINL void load_b_frag(Frag& f, const half_t* p, int n0, int k0, int stride, int lane) {
  const int nc = lane & 15, g = lane >> 4;
  const half_t* base = p + (n0 + nc) * stride + k0 + 8 * g;
  f.q[0] = *reinterpret_cast<const uint4*>(base);
  f.q[1] = *reinterpret_cast<const uint4*>(base + 16);
}

DEVINL v8f wmma_f16(const Frag& a, const Frag& b, v8f c) {
  return __builtin_amdgcn_wmma_f32_16x16x32_f16(false, a.h, false, b.h,
                                                (short)0, c, false, false);
}

// Software-pipelined LDS-operand GEMM: NJ 16-col tiles x NKS 32-deep k-steps.
// acols[ks] gives the physical A-tile column base per logical k-step (handles
// non-contiguous feature concatenations). Double-buffers A and B fragments so
// every WMMA only waits on loads issued one stage earlier.
template <int NJ, int NKS>
DEVINL void gemm_lds(v8f* acc, const half_t* A, int arow, const int (&acols)[NKS],
                     int astride, const half_t* Bw, int bstride, int lane) {
  Frag a[2], bf[2];
  load_a_frag(a[0], A, arow, acols[0], astride, lane);
  load_b_frag(bf[0], Bw, 0, 0, bstride, lane);
#pragma unroll
  for (int ks = 0; ks < NKS; ++ks) {
#pragma unroll
    for (int j = 0; j < NJ; ++j) {
      if (j < NJ - 1) {
        load_b_frag(bf[(j + 1) & 1], Bw, (j + 1) * 16, ks * 32, bstride, lane);
      } else if (ks < NKS - 1) {
        load_a_frag(a[(ks + 1) & 1], A, arow, acols[ks + 1], astride, lane);
        load_b_frag(bf[(j + 1) & 1], Bw, 0, (ks + 1) * 32, bstride, lane);
      }
      acc[j] = wmma_f16(a[ks & 1], bf[j & 1], acc[j]);
    }
  }
}

union H4 { half_t h[4]; uint2 u; };

// ---------------- main fused kernel ----------------------------------------
__global__ __launch_bounds__(256) void gnn_fused(
    const float* __restrict__ in_nodes,   // [B][N][64]
    const float* __restrict__ in_edges,   // [B][E][64]
    const half_t* __restrict__ wg,        // f16 weight arena (d_ws)
    const float* __restrict__ bn1, const float* __restrict__ bn2,
    const float* __restrict__ be1, const float* __restrict__ be2,
    const float* __restrict__ bo,
    float* __restrict__ out_nodes,        // [B][N][64]
    float* __restrict__ out_edges) {      // [B][E][64]
  extern __shared__ char smem[];
  half_t* lw   = reinterpret_cast<half_t*>(smem + LB_W);
  half_t* xn   = reinterpret_cast<half_t*>(smem + LB_XN);  // 128 x [src|dst|edge]
  half_t* Hl   = reinterpret_cast<half_t*>(smem + LB_H);   // 128 x 128 hidden
  float*  PEl  = reinterpret_cast<float*>(smem + LB_PE);   // pair_e f32
  float*  red  = reinterpret_cast<float*>(smem + LB_RED);  // [16][64] partials
  float*  agg  = reinterpret_cast<float*>(smem + LB_AGG);  // [2][64]
  float*  sb1n = reinterpret_cast<float*>(smem + LB_B1N);
  float*  sb1e = reinterpret_cast<float*>(smem + LB_B1E);
  float*  sb2n = reinterpret_cast<float*>(smem + LB_B2N);
  float*  sb2e = reinterpret_cast<float*>(smem + LB_B2E);
  float*  sbo  = reinterpret_cast<float*>(smem + LB_BO);

  const int tid  = threadIdx.x;
  const int lane = tid & 31;
  const int wid  = tid >> 5;      // 8 row-waves, rows [16*wid, 16*wid+16)
  const int bw   = wid >> 2;      // batch-half of this wave (0 or 1)
  const int wml  = wid & 3;       // row-wave within its batch
  const int gg   = lane >> 4;
  const int c0   = lane & 15;

  const v8f vzero = {0.f, 0.f, 0.f, 0.f, 0.f, 0.f, 0.f, 0.f};

  // physical A-column base per logical k-step for each GEMM
  const int colsN1[6] = {0, 32, 64, 96, 128, 160};  // [src|dst|edge]
  const int colsE1[4] = {0, 32, 128, 160};          // [src|edge]
  const int colsH[4]  = {0, 32, 64, 96};            // hidden
  const int cols3[4]  = {64, 96, 128, 160};         // [mean_e|edge]

  // ---- cache f16 weights + biases in LDS (amortized over the tile loop) ---
  {
    const uint4* s = reinterpret_cast<const uint4*>(wg);
    uint4* d = reinterpret_cast<uint4*>(lw);
    for (int i = tid; i < W_HALVES / 8; i += 256) d[i] = s[i];
    if (tid < 128) { sb1n[tid] = bn1[tid]; sb1e[tid] = be1[tid]; }
    if (tid < 64)  { sb2n[tid] = bn2[tid]; sb2e[tid] = be2[tid]; sbo[tid] = bo[tid]; }
  }
  __syncthreads();

  // one tile = one node n for a pair of batches (b0, b0+1)
  for (int t = blockIdx.x; t < (Bc / 2) * Nn; t += gridDim.x) {
    const int b0 = (t >> 10) * 2;
    const int n = t & 1023;
    const int w = (n < 64) ? n : 64;
    const long long start = (n <= 64) ? (long long)n * (n - 1) / 2
                                      : 2016LL + (long long)(n - 64) * 64;

    // ---- zero A-tile (b128 stores) ---------------------------------------
    {
      const uint4 z4 = {0u, 0u, 0u, 0u};
      uint4* xq = reinterpret_cast<uint4*>(xn);
      for (int i = tid; i < 128 * SXN / 8; i += 256) xq[i] = z4;
    }
    __syncthreads();

    // ---- stage [src | dst | edge] rows r < w for both batches as f16 -----
    for (int i = tid; i < 2 * w * 48; i += 256) {
      const int r2 = i / 48, s = i % 48;
      const int bb = (r2 >= w) ? 1 : 0;
      const int r = r2 - bb * w;
      const float* nodes_b = in_nodes + (long long)(b0 + bb) * Nn * FN;
      const float4* p;
      int colb;
      if (s < 16) {        // nodes_src: contiguous rows [n-w, n)
        p = reinterpret_cast<const float4*>(nodes_b + (long long)(n - w + r) * FN) + s;
        colb = s * 4;
      } else if (s < 32) { // nodes_dst: broadcast row n
        p = reinterpret_cast<const float4*>(nodes_b + (long long)n * FN) + (s - 16);
        colb = 64 + (s - 16) * 4;
      } else {             // edges: contiguous rows [start, start+w)
        p = reinterpret_cast<const float4*>(
                in_edges + ((long long)(b0 + bb) * E_TOT + start + r) * 64) + (s - 32);
        colb = 128 + (s - 32) * 4;
      }
      const float4 v = *p;
      H4 t4;
      t4.h[0] = (half_t)v.x; t4.h[1] = (half_t)v.y;
      t4.h[2] = (half_t)v.z; t4.h[3] = (half_t)v.w;
      *reinterpret_cast<uint2*>(xn + (bb * 64 + r) * SXN + colb) = t4.u;
    }
    __syncthreads();

    // ---- node path GEMM1: (128x192)@(192x128) -> H -----------------------
    v8f acc1[8];
#pragma unroll
    for (int j = 0; j < 8; ++j) acc1[j] = vzero;
    gemm_lds<8, 6>(acc1, xn, wid * 16, colsN1, SXN, lw + OFF_WN1, SW1, lane);
#pragma unroll
    for (int j = 0; j < 8; ++j) {
      const int col = j * 16 + c0;
      const float bb = sb1n[col];
#pragma unroll
      for (int v = 0; v < 8; ++v)
        Hl[(wid * 16 + v + 8 * gg) * SH + col] = (half_t)(acc1[j][v] + bb);
    }
    __syncthreads();

    // ---- node path GEMM2: (128x128)@(128x64) -> pair_n, relu, colsum -----
    v8f acc2[4];
#pragma unroll
    for (int j = 0; j < 4; ++j) acc2[j] = vzero;
    gemm_lds<4, 4>(acc2, Hl, wid * 16, colsH, SH, lw + OFF_WN2, SW, lane);
#pragma unroll
    for (int j = 0; j < 4; ++j) {
      const int col = j * 16 + c0;
      const float bb = sb2n[col];
      float part = 0.f;
#pragma unroll
      for (int v = 0; v < 8; ++v) {
        const int ml = wml * 16 + v + 8 * gg;  // row within this batch
        float pv = acc2[j][v] + bb;
        pv = pv > 0.f ? pv : 0.f;
        if (ml < w) part += pv;
      }
      red[(wid * 2 + gg) * 64 + col] = part;  // fixed-order -> deterministic
    }
    __syncthreads();

    // ---- edge path GEMM1: (128x[src|edge])@(128x128) -> H ----------------
#pragma unroll
    for (int j = 0; j < 8; ++j) acc1[j] = vzero;
    gemm_lds<8, 4>(acc1, xn, wid * 16, colsE1, SXN, lw + OFF_WE1, SW, lane);
#pragma unroll
    for (int j = 0; j < 8; ++j) {
      const int col = j * 16 + c0;
      const float bb = sb1e[col];
#pragma unroll
      for (int v = 0; v < 8; ++v)
        Hl[(wid * 16 + v + 8 * gg) * SH + col] = (half_t)(acc1[j][v] + bb);
    }
    __syncthreads();

    // ---- edge path GEMM2 -> pair_e (relu) into LDS f32 -------------------
    v8f acc2e[4];
#pragma unroll
    for (int j = 0; j < 4; ++j) acc2e[j] = vzero;
    gemm_lds<4, 4>(acc2e, Hl, wid * 16, colsH, SH, lw + OFF_WE2, SW, lane);
#pragma unroll
    for (int j = 0; j < 4; ++j) {
      const int col = j * 16 + c0;
      const float bb = sb2e[col];
#pragma unroll
      for (int v = 0; v < 8; ++v) {
        float pv = acc2e[j][v] + bb;
        PEl[(wid * 16 + v + 8 * gg) * SPE + col] = pv > 0.f ? pv : 0.f;
      }
    }
    __syncthreads();

    // ---- exclusive prefix mean -> mean_e (f16 into xn cols [64,128)),
    // ---- agg_n = fixed-order sum of 8 partials per batch ------------------
    if (tid < 128) {            // 2 batches x 64 columns scan in parallel
      const int bb = tid >> 6, c = tid & 63;
      float run = 0.f;
      for (int r = 0; r < w; ++r) {
        const float mv = (r == 0) ? 0.f : run / (float)r;
        xn[(bb * 64 + r) * SXN + 64 + c] = (half_t)mv;
        run += PEl[(bb * 64 + r) * SPE + c];
      }
    } else {
      const int idx = tid - 128;  // 2 batches x 64 columns
      const int bb = idx >> 6, c = idx & 63;
      float s = 0.f;
#pragma unroll
      for (int i = 0; i < 8; ++i) s += red[(bb * 8 + i) * 64 + c];
      agg[idx] = s;
    }
    __syncthreads();

    // ---- GEMM3: (128x[mean_e|edge])@(128x64) + bias, relu -> out_edges ---
    v8f acc3[4];
#pragma unroll
    for (int j = 0; j < 4; ++j) acc3[j] = vzero;
    gemm_lds<4, 4>(acc3, xn, wid * 16, cols3, SXN, lw + OFF_WEO, SW, lane);
    {
      float* oe = out_edges + ((long long)(b0 + bw) * E_TOT + start) * 64;
#pragma unroll
      for (int j = 0; j < 4; ++j) {
        const int col = j * 16 + c0;
        const float bb = sbo[col];
#pragma unroll
        for (int v = 0; v < 8; ++v) {
          const int ml = wml * 16 + v + 8 * gg;
          if (ml < w) {
            float pv = acc3[j][v] + bb;
            oe[(long long)ml * 64 + col] = pv > 0.f ? pv : 0.f;
          }
        }
      }
    }

    // ---- out_nodes rows: [agg/max(w,1) | node_n] @ W_nodes + bias, relu --
    if (tid < 128) {
      const int bb = tid >> 6, o = tid & 63;
      const float invw = 1.f / (float)(w > 0 ? w : 1);
      const half_t* Wt = lw + OFF_WNO + o * SW;  // [o][k]
      float a = sbo[o];
      const float* nr = in_nodes + ((long long)(b0 + bb) * Nn + n) * FN;
      for (int k = 0; k < 64; ++k) a += (agg[bb * 64 + k] * invw) * (float)Wt[k];
      for (int k = 0; k < 64; ++k) a += nr[k] * (float)Wt[64 + k];
      out_nodes[((long long)(b0 + bb) * Nn + n) * 64 + o] = a > 0.f ? a : 0.f;
    }
    __syncthreads();  // protect xn/Hl/PEl/red/agg before next tile
  }
}

// ---------------- host entry ------------------------------------------------
extern "C" void kernel_launch(void* const* d_in, const int* in_sizes, int n_in,
                              void* d_out, int out_size, void* d_ws, size_t ws_size,
                              hipStream_t stream) {
  (void)in_sizes; (void)n_in; (void)out_size; (void)ws_size;
  const float* in_nodes = (const float*)d_in[0];
  const float* in_edges = (const float*)d_in[1];
  const float* Wn1 = (const float*)d_in[2];
  const float* bn1 = (const float*)d_in[3];
  const float* Wn2 = (const float*)d_in[4];
  const float* bn2 = (const float*)d_in[5];
  const float* We1 = (const float*)d_in[6];
  const float* be1 = (const float*)d_in[7];
  const float* We2 = (const float*)d_in[8];
  const float* be2 = (const float*)d_in[9];
  const float* Wno = (const float*)d_in[10];
  const float* Weo = (const float*)d_in[11];
  const float* bo  = (const float*)d_in[12];

  half_t* ws = (half_t*)d_ws;

  // prologue: transpose-convert all weights to f16 [N][K+pad] in d_ws
  struct Cvt { const float* s; int off, K, N, st; } cv[6] = {
      {Wn1, OFF_WN1, 192, 128, SW1}, {Wn2, OFF_WN2, 128, 64, SW},
      {We1, OFF_WE1, 128, 128, SW},  {We2, OFF_WE2, 128, 64, SW},
      {Weo, OFF_WEO, 128, 64, SW},   {Wno, OFF_WNO, 128, 64, SW}};
  for (int i = 0; i < 6; ++i) {
    const int tot = cv[i].N * cv[i].st;
    wcvt_kernel<<<(tot + 255) / 256, 256, 0, stream>>>(cv[i].s, ws + cv[i].off,
                                                       cv[i].K, cv[i].N, cv[i].st);
  }

  float* out_nodes = (float*)d_out;
  float* out_edges = out_nodes + (size_t)Bc * Nn * 64;

  gnn_fused<<<1024, 256, LDS_BYTES, stream>>>(
      in_nodes, in_edges, ws, bn1, bn2, be1, be2, bo, out_nodes, out_edges);
}
